// PointNet2MSG_55216099557664
// MI455X (gfx1250) — matrix operand order repaired and checked by reading, not compiled, not run on hardware
//
#include <hip/hip_runtime.h>
#include <hip/hip_bf16.h>

typedef __attribute__((ext_vector_type(16))) _Float16 v16h;
typedef __attribute__((ext_vector_type(8)))  _Float16 v8h;
typedef __attribute__((ext_vector_type(8)))  float    v8f;

#define BN_EPS 1e-5f

// ---------------------------------------------------------------------------
// Small prep kernels
// ---------------------------------------------------------------------------
__global__ void f32_to_f16_kernel(const float* __restrict__ in, _Float16* __restrict__ out, int n) {
  int i = blockIdx.x * blockDim.x + threadIdx.x;
  if (i < n) out[i] = (_Float16)in[i];
}

__global__ void zero_f16_kernel(_Float16* __restrict__ p, int n) {
  int i = blockIdx.x * blockDim.x + threadIdx.x;
  if (i < n) p[i] = (_Float16)0.f;
}

// Pack [cin,cout] f32 weights into a WMMA-ready swizzled f16 layout:
//   dst[((k>>5)*cout + c)*32 + (k&31)] = w[k*cout + c]   (zero for k >= cin)
// so a lane (column c, K-block k0, lane-half kbB) reads halves kbB+0..7 and kbB+16..23 (two 8-half runs).
__global__ void pack_w_kernel(const float* __restrict__ w, int cin, int cout, int kp,
                              _Float16* __restrict__ out) {
  int i = blockIdx.x * blockDim.x + threadIdx.x;
  if (i >= kp * cout) return;
  int k = i / cout, c = i % cout;
  _Float16 v = (k < cin) ? (_Float16)w[k * cout + c] : (_Float16)0.f;
  out[(((size_t)(k >> 5) * cout + c) << 5) + (k & 31)] = v;
}

// fold BN into scale/bias: sb[0..c) = s, sb[c..2c) = t  where y = acc*s + t
__global__ void bn_fold_kernel(const float* __restrict__ g, const float* __restrict__ b,
                               const float* __restrict__ m, const float* __restrict__ v,
                               int c, float* __restrict__ sb) {
  int i = blockIdx.x * blockDim.x + threadIdx.x;
  if (i >= c) return;
  float s = g[i] * rsqrtf(v[i] + BN_EPS);
  sb[i] = s;
  sb[c + i] = b[i] - m[i] * s;
}

// write xyz2 into cols 0..2 of the SA3 group-all panel [B,128,672]
__global__ void xyz_to_sa3_kernel(const float* __restrict__ xyz2, _Float16* __restrict__ sa3in) {
  int i = blockIdx.x * blockDim.x + threadIdx.x;
  if (i >= 32 * 128 * 3) return;
  int bp = i / 3, c = i % 3;
  sa3in[(size_t)bp * 672 + c] = (_Float16)xyz2[i];
}

// ---------------------------------------------------------------------------
// Farthest point sampling: one block per batch, strided per-thread distances
// ---------------------------------------------------------------------------
template <int N, int NPT>
__global__ __launch_bounds__(256) void fps_kernel(const float* __restrict__ xyz, int P,
                                                  int* __restrict__ out_idx,
                                                  float* __restrict__ out_xyz) {
  const int b = blockIdx.x, tid = threadIdx.x;
  const float* X = xyz + (size_t)b * N * 3;
  __shared__ float lastP[3];
  __shared__ float rv[256];
  __shared__ int ri[256];
  float d[NPT];
#pragma unroll
  for (int j = 0; j < NPT; ++j) d[j] = 1e10f;
  if (tid == 0) {
    out_idx[b * P] = 0;
    lastP[0] = X[0]; lastP[1] = X[1]; lastP[2] = X[2];
    out_xyz[(size_t)b * P * 3 + 0] = X[0];
    out_xyz[(size_t)b * P * 3 + 1] = X[1];
    out_xyz[(size_t)b * P * 3 + 2] = X[2];
  }
  __syncthreads();
  for (int it = 1; it < P; ++it) {
    float lx = lastP[0], ly = lastP[1], lz = lastP[2];
    float best = -1.f; int bi = tid;
#pragma unroll
    for (int j = 0; j < NPT; ++j) {
      int i = tid + j * 256;
      float dx = X[i * 3] - lx, dy = X[i * 3 + 1] - ly, dz = X[i * 3 + 2] - lz;
      float dd = dx * dx + dy * dy + dz * dz;
      d[j] = fminf(d[j], dd);
      if (d[j] > best) { best = d[j]; bi = i; }
    }
    rv[tid] = best; ri[tid] = bi;
    __syncthreads();
    for (int s = 128; s > 0; s >>= 1) {
      if (tid < s && rv[tid + s] > rv[tid]) { rv[tid] = rv[tid + s]; ri[tid] = ri[tid + s]; }
      __syncthreads();
    }
    if (tid == 0) {
      int sel = ri[0];
      out_idx[b * P + it] = sel;
      lastP[0] = X[sel * 3]; lastP[1] = X[sel * 3 + 1]; lastP[2] = X[sel * 3 + 2];
      out_xyz[((size_t)b * P + it) * 3 + 0] = lastP[0];
      out_xyz[((size_t)b * P + it) * 3 + 1] = lastP[1];
      out_xyz[((size_t)b * P + it) * 3 + 2] = lastP[2];
    }
    __syncthreads();
  }
}

// ---------------------------------------------------------------------------
// Ball query: one wave32 per query point; first NS in-ball in index order
// ---------------------------------------------------------------------------
__global__ __launch_bounds__(32) void ball_query_kernel(const float* __restrict__ src_xyz, int Nsrc,
                                                        const float* __restrict__ centers,
                                                        float r2, int NS,
                                                        int* __restrict__ out, int P) {
  const int bp = blockIdx.x;
  const int b = bp / P;
  const int lane = threadIdx.x;
  const float* X = src_xyz + (size_t)b * Nsrc * 3;
  const float cx = centers[(size_t)bp * 3 + 0];
  const float cy = centers[(size_t)bp * 3 + 1];
  const float cz = centers[(size_t)bp * 3 + 2];
  __shared__ int cnt;
  if (lane == 0) cnt = 0;
  __syncthreads();
  for (int base = 0; base < Nsrc; base += 32) {
    int c0 = cnt;
    if (c0 >= NS) break;
    int i = base + lane;
    bool in = false;
    if (i < Nsrc) {
      float dx = X[i * 3] - cx, dy = X[i * 3 + 1] - cy, dz = X[i * 3 + 2] - cz;
      in = (dx * dx + dy * dy + dz * dz) < r2;
    }
    unsigned mask = (unsigned)__ballot(in);
    int pre = __popc(mask & ((1u << lane) - 1u));
    if (in && (c0 + pre) < NS) out[(size_t)bp * NS + c0 + pre] = i;
    __syncthreads();
    if (lane == 0) cnt = c0 + __popc(mask);
    __syncthreads();
  }
  __syncthreads();
  int c = cnt < NS ? cnt : NS;
  int first = out[(size_t)bp * NS];  // center itself is always in-ball -> c >= 1
  for (int j = c + lane; j < NS; j += 32) out[(size_t)bp * NS + j] = first;
}

// ---------------------------------------------------------------------------
// WMMA GEMM over an LDS activation panel.
//   A: LDS, R x K (padded stride AS), f16. Wp: global, swizzled K-packed f16.
//   sb: [s(N) || t(N)].  POOL=false: Dst = relu(acc*s+t) as f16 (stride DS).
//   POOL=true: shared float-bit atomicMax pool (values >= 0 after ReLU).
// Steady-state inner loop: 2x ds_load_b128 (A) + 2x global_load_b128 (B)
// + shuffles + v_wmma_f32_16x16x32_f16. Wave-uniform control flow.
// ---------------------------------------------------------------------------
template <int R, int K, int N, int AS, int DS, bool POOL>
__device__ __forceinline__ void gemm_panel(const _Float16* __restrict__ A,
                                           const _Float16* __restrict__ Wp,
                                           const float* __restrict__ sb,
                                           _Float16* __restrict__ Dst,
                                           int* __restrict__ pool) {
  static_assert(R % 16 == 0 && K % 32 == 0 && N % 16 == 0, "tile shape");
  const int wave = threadIdx.x >> 5;
  const int nWaves = blockDim.x >> 5;
  const int lane = threadIdx.x & 31;
  const int l16 = lane & 15;
  const int hi = (lane & 16) ? 1 : 0;  // lane half
  const int kbA = hi ? 8 : 0;          // A: K base offset (dense 16x32 f16 layout)
  const int kbB = hi ? 8 : 0;          // B: K base offset (dense 16x32 f16 layout)
  constexpr int MT = R / 16, NT = N / 16;
  for (int t = wave; t < MT * NT; t += nWaves) {
    const int mt = t / NT, nt = t % NT;
    const int arow = mt * 16 + l16;
    const int bcol = nt * 16 + l16;
    const _Float16* ap = A + arow * AS + kbA;            // 16B aligned
    const _Float16* wptr = Wp + ((size_t)bcol << 5) + kbB;  // 16B aligned
    const size_t wstep = (size_t)N << 5;
    v8f acc = {};
    for (int k0 = 0; k0 < K; k0 += 32) {
      v8h a0 = *(const v8h*)(ap);        // K = kbA + 0..7
      v8h a1 = *(const v8h*)(ap + 16);   // K = kbA + 16..23
      v8h w0 = *(const v8h*)(wptr);      // K = kbB + 0..7
      v8h w1 = *(const v8h*)(wptr + 16); // K = kbB + 16..23
      __builtin_prefetch(wptr + wstep, 0, 3);
      v16h av = __builtin_shufflevector(a0, a1, 0, 1, 2, 3, 4, 5, 6, 7,
                                        8, 9, 10, 11, 12, 13, 14, 15);
      v16h bv = __builtin_shufflevector(w0, w1, 0, 1, 2, 3, 4, 5, 6, 7,
                                        8, 9, 10, 11, 12, 13, 14, 15);
      acc = __builtin_amdgcn_wmma_f32_16x16x32_f16(false, av, false, bv,
                                                   (short)0, acc, false, false);
      ap += 32;
      wptr += wstep;
    }
#pragma unroll
    for (int r = 0; r < 8; ++r) {
      const int orow = mt * 16 + r + (hi ? 8 : 0);
      const int ocol = nt * 16 + l16;
      float y = acc[r] * sb[ocol] + sb[N + ocol];
      y = fmaxf(y, 0.f);
      if constexpr (POOL) {
        atomicMax(&pool[ocol], __float_as_int(y));  // y >= 0 -> int max == float max
      } else {
        Dst[orow * DS + ocol] = (_Float16)y;
      }
    }
  }
}

// ---------------------------------------------------------------------------
// Fused SA stage: one block per (b, point). Gather neighbor chunk -> LDS,
// run 3 WMMA MLP layers, stream last layer into max-pool, emit f16 features.
// ---------------------------------------------------------------------------
template <int NS, int CIN, int KP, int C1, int C2, int C3, int R>
__global__ __launch_bounds__(128) void sa_fused_kernel(
    const float* __restrict__ src_xyz, int Nsrc,
    const _Float16* __restrict__ src_feat, int C,
    const int* __restrict__ ball_idx,
    const float* __restrict__ centers,
    const _Float16* __restrict__ W1, const _Float16* __restrict__ W2,
    const _Float16* __restrict__ W3,
    const float* __restrict__ sb1, const float* __restrict__ sb2,
    const float* __restrict__ sb3,
    _Float16* __restrict__ out16, int out_stride, int out_off, int P) {
  static_assert(NS % R == 0, "chunking");
  constexpr int PA = ((KP > C2) ? KP : C2) + 8;  // +8 halves: odd 16B stride
  constexpr int PB = C1 + 8;
  __shared__ _Float16 pa[R * PA];
  __shared__ _Float16 pb[R * PB];
  __shared__ int pool[C3];
  __shared__ int sidx[NS];
  __shared__ float ctr[3];
  const int bp = blockIdx.x;
  const int b = bp / P;
  for (int t = threadIdx.x; t < NS; t += blockDim.x)
    sidx[t] = ball_idx[(size_t)bp * NS + t];
  for (int t = threadIdx.x; t < C3; t += blockDim.x) pool[t] = 0;
  if (threadIdx.x < 3) ctr[threadIdx.x] = centers[(size_t)bp * 3 + threadIdx.x];
  __syncthreads();
  for (int s0 = 0; s0 < NS; s0 += R) {
    // build grouped input panel: [centered xyz || feats || zero pad]
    for (int t = threadIdx.x; t < R * KP; t += blockDim.x) {
      int r = t / KP, c = t % KP;
      int gi = sidx[s0 + r];
      _Float16 v = (_Float16)0.f;
      if (c < 3)
        v = (_Float16)(src_xyz[((size_t)b * Nsrc + gi) * 3 + c] - ctr[c]);
      else if (c < CIN)
        v = src_feat[((size_t)b * Nsrc + gi) * C + (c - 3)];
      pa[r * PA + c] = v;
    }
    __syncthreads();
    gemm_panel<R, KP, C1, PA, PB, false>(pa, W1, sb1, pb, nullptr);
    __syncthreads();
    gemm_panel<R, C1, C2, PB, PA, false>(pb, W2, sb2, pa, nullptr);
    __syncthreads();
    gemm_panel<R, C2, C3, PA, 1, true>(pa, W3, sb3, nullptr, pool);
    __syncthreads();
  }
  for (int t = threadIdx.x; t < C3; t += blockDim.x)
    out16[(size_t)bp * out_stride + out_off + t] = (_Float16)__int_as_float(pool[t]);
}

// ---------------------------------------------------------------------------
// Fused SA3 (group-all): one block per batch over prebuilt [128 x KP] panel
// ---------------------------------------------------------------------------
template <int KP, int C1, int C2, int C3, int R>
__global__ __launch_bounds__(128) void sa3_fused_kernel(
    const _Float16* __restrict__ inb,  // [B,128,KP]
    const _Float16* __restrict__ W1, const _Float16* __restrict__ W2,
    const _Float16* __restrict__ W3,
    const float* __restrict__ sb1, const float* __restrict__ sb2,
    const float* __restrict__ sb3,
    float* __restrict__ outf) {  // [B, C3]
  constexpr int PA = ((KP > C2) ? KP : C2) + 8;
  constexpr int PB = C1 + 8;
  __shared__ _Float16 pa[R * PA];
  __shared__ _Float16 pb[R * PB];
  __shared__ int pool[C3];
  const int b = blockIdx.x;
  for (int t = threadIdx.x; t < C3; t += blockDim.x) pool[t] = 0;
  __syncthreads();
  for (int s0 = 0; s0 < 128; s0 += R) {
    for (int t = threadIdx.x; t < R * KP; t += blockDim.x) {
      int r = t / KP, c = t % KP;
      pa[r * PA + c] = inb[((size_t)b * 128 + s0 + r) * KP + c];
    }
    __syncthreads();
    gemm_panel<R, KP, C1, PA, PB, false>(pa, W1, sb1, pb, nullptr);
    __syncthreads();
    gemm_panel<R, C1, C2, PB, PA, false>(pb, W2, sb2, pa, nullptr);
    __syncthreads();
    gemm_panel<R, C2, C3, PA, 1, true>(pa, W3, sb3, nullptr, pool);
    __syncthreads();
  }
  for (int t = threadIdx.x; t < C3; t += blockDim.x)
    outf[(size_t)b * C3 + t] = __int_as_float(pool[t]);
}

// ---------------------------------------------------------------------------
// FC head (tiny FLOPs): y = act_bn(x @ w (+ lb)); slope 1.0 => linear
// ---------------------------------------------------------------------------
__global__ void head_kernel(const float* __restrict__ x, const float* __restrict__ w,
                            const float* __restrict__ lb,
                            const float* __restrict__ g, const float* __restrict__ bb,
                            const float* __restrict__ mm, const float* __restrict__ vv,
                            float slope, float* __restrict__ y,
                            int rows, int cin, int cout) {
  int i = blockIdx.x * blockDim.x + threadIdx.x;
  if (i >= rows * cout) return;
  int r = i / cout, c = i % cout;
  float acc = 0.f;
  for (int k = 0; k < cin; ++k) acc += x[(size_t)r * cin + k] * w[(size_t)k * cout + c];
  if (lb) acc += lb[c];
  if (g) {
    float s = g[c] * rsqrtf(vv[c] + BN_EPS);
    acc = (acc - mm[c]) * s + bb[c];
  }
  if (acc < 0.f) acc *= slope;
  y[i] = acc;
}

// ---------------------------------------------------------------------------
// Host orchestration
// ---------------------------------------------------------------------------
extern "C" void kernel_launch(void* const* d_in, const int* in_sizes, int n_in,
                              void* d_out, int out_size, void* d_ws, size_t ws_size,
                              hipStream_t stream) {
  (void)in_sizes; (void)n_in; (void)out_size; (void)ws_size;
  const int B = 32, N = 4096, P1 = 512, P2 = 128;
  const float* xyz = (const float*)d_in[0];
  const float* rgb = (const float*)d_in[1];
  auto F = [&](int i) { return (const float*)d_in[i]; };

  // ---- workspace carving (256B aligned) ----
  char* ws = (char*)d_ws;
  size_t off = 0;
  auto alloc = [&](size_t bytes) -> void* {
    void* p = ws + off;
    off = (off + bytes + 255) & ~(size_t)255;
    return p;
  };
  float* xyz1 = (float*)alloc((size_t)B * P1 * 3 * 4);
  int* idx1 = (int*)alloc((size_t)B * P1 * 4);
  float* xyz2 = (float*)alloc((size_t)B * P2 * 3 * 4);
  int* idx2 = (int*)alloc((size_t)B * P2 * 4);
  _Float16* rgbh = (_Float16*)alloc((size_t)B * N * 3 * 2);
  int* ball1[3] = {(int*)alloc((size_t)B * P1 * 16 * 4),
                   (int*)alloc((size_t)B * P1 * 32 * 4),
                   (int*)alloc((size_t)B * P1 * 128 * 4)};
  int* ball2[3] = {(int*)alloc((size_t)B * P2 * 32 * 4),
                   (int*)alloc((size_t)B * P2 * 64 * 4),
                   (int*)alloc((size_t)B * P2 * 128 * 4)};
  _Float16* feats1 = (_Float16*)alloc((size_t)B * P1 * 320 * 2);
  _Float16* sa3in = (_Float16*)alloc((size_t)B * P2 * 672 * 2);
  float* feat = (float*)alloc((size_t)B * 1024 * 4);
  float* h1 = (float*)alloc((size_t)B * 512 * 4);
  float* h2 = (float*)alloc((size_t)B * 512 * 4);

  // ---- layer table: {param base idx, cin, cout, padded K} ----
  struct LayerDef { int base, cin, cout, kp; };
  static const LayerDef L[21] = {
      {2, 6, 32, 32},    {7, 32, 32, 32},   {12, 32, 64, 32},     // sa1 s0
      {17, 6, 64, 32},   {22, 64, 64, 64},  {27, 64, 128, 64},    // sa1 s1
      {32, 6, 64, 32},   {37, 64, 96, 64},  {42, 96, 128, 96},    // sa1 s2
      {47, 323, 64, 352},{52, 64, 64, 64},  {57, 64, 128, 64},    // sa2 s0
      {62, 323, 128, 352},{67, 128, 128, 128},{72, 128, 256, 128},// sa2 s1
      {77, 323, 128, 352},{82, 128, 128, 128},{87, 128, 256, 128},// sa2 s2
      {92, 643, 256, 672},{97, 256, 512, 256},{102, 512, 1024, 512}}; // sa3
  _Float16* wp[21];
  float* sbp[21];
  for (int l = 0; l < 21; ++l) {
    wp[l] = (_Float16*)alloc((size_t)L[l].kp * L[l].cout * 2);
    sbp[l] = (float*)alloc((size_t)2 * L[l].cout * 4);
  }
  for (int l = 0; l < 21; ++l) {
    int n = L[l].kp * L[l].cout;
    pack_w_kernel<<<(n + 255) / 256, 256, 0, stream>>>(F(L[l].base), L[l].cin, L[l].cout,
                                                       L[l].kp, wp[l]);
    bn_fold_kernel<<<(L[l].cout + 255) / 256, 256, 0, stream>>>(
        F(L[l].base + 1), F(L[l].base + 2), F(L[l].base + 3), F(L[l].base + 4),
        L[l].cout, sbp[l]);
  }

  // ---- stage inputs ----
  f32_to_f16_kernel<<<(B * N * 3 + 255) / 256, 256, 0, stream>>>(rgb, rgbh, B * N * 3);

  // ---- SA1 ----
  fps_kernel<4096, 16><<<B, 256, 0, stream>>>(xyz, P1, idx1, xyz1);
  ball_query_kernel<<<B * P1, 32, 0, stream>>>(xyz, N, xyz1, 0.01f, 16, ball1[0], P1);
  ball_query_kernel<<<B * P1, 32, 0, stream>>>(xyz, N, xyz1, 0.04f, 32, ball1[1], P1);
  ball_query_kernel<<<B * P1, 32, 0, stream>>>(xyz, N, xyz1, 0.16f, 128, ball1[2], P1);
  sa_fused_kernel<16, 6, 32, 32, 32, 64, 16><<<B * P1, 128, 0, stream>>>(
      xyz, N, rgbh, 3, ball1[0], xyz1, wp[0], wp[1], wp[2], sbp[0], sbp[1], sbp[2],
      feats1, 320, 0, P1);
  sa_fused_kernel<32, 6, 32, 64, 64, 128, 32><<<B * P1, 128, 0, stream>>>(
      xyz, N, rgbh, 3, ball1[1], xyz1, wp[3], wp[4], wp[5], sbp[3], sbp[4], sbp[5],
      feats1, 320, 64, P1);
  sa_fused_kernel<128, 6, 32, 64, 96, 128, 32><<<B * P1, 128, 0, stream>>>(
      xyz, N, rgbh, 3, ball1[2], xyz1, wp[6], wp[7], wp[8], sbp[6], sbp[7], sbp[8],
      feats1, 320, 192, P1);

  // ---- SA2 ----
  fps_kernel<512, 2><<<B, 256, 0, stream>>>(xyz1, P2, idx2, xyz2);
  ball_query_kernel<<<B * P2, 32, 0, stream>>>(xyz1, P1, xyz2, 0.04f, 32, ball2[0], P2);
  ball_query_kernel<<<B * P2, 32, 0, stream>>>(xyz1, P1, xyz2, 0.16f, 64, ball2[1], P2);
  ball_query_kernel<<<B * P2, 32, 0, stream>>>(xyz1, P1, xyz2, 0.64f, 128, ball2[2], P2);
  zero_f16_kernel<<<(B * P2 * 672 + 255) / 256, 256, 0, stream>>>(sa3in, B * P2 * 672);
  xyz_to_sa3_kernel<<<(B * P2 * 3 + 255) / 256, 256, 0, stream>>>(xyz2, sa3in);
  sa_fused_kernel<32, 323, 352, 64, 64, 128, 32><<<B * P2, 128, 0, stream>>>(
      xyz1, P1, feats1, 320, ball2[0], xyz2, wp[9], wp[10], wp[11], sbp[9], sbp[10],
      sbp[11], sa3in, 672, 3, P2);
  sa_fused_kernel<64, 323, 352, 128, 128, 256, 32><<<B * P2, 128, 0, stream>>>(
      xyz1, P1, feats1, 320, ball2[1], xyz2, wp[12], wp[13], wp[14], sbp[12], sbp[13],
      sbp[14], sa3in, 672, 3 + 128, P2);
  sa_fused_kernel<128, 323, 352, 128, 128, 256, 32><<<B * P2, 128, 0, stream>>>(
      xyz1, P1, feats1, 320, ball2[2], xyz2, wp[15], wp[16], wp[17], sbp[15], sbp[16],
      sbp[17], sa3in, 672, 3 + 384, P2);

  // ---- SA3 (group-all) ----
  sa3_fused_kernel<672, 256, 512, 1024, 16><<<B, 128, 0, stream>>>(
      sa3in, wp[18], wp[19], wp[20], sbp[18], sbp[19], sbp[20], feat);

  // ---- head: fc1 (BN + leaky 0.2), fc2 (BN, linear), cls ----
  head_kernel<<<(B * 512 + 255) / 256, 256, 0, stream>>>(
      feat, F(107), F(108), F(109), F(110), F(111), F(112), 0.2f, h1, B, 1024, 512);
  head_kernel<<<(B * 512 + 255) / 256, 256, 0, stream>>>(
      h1, F(113), F(114), F(115), F(116), F(117), F(118), 1.0f, h2, B, 512, 512);
  head_kernel<<<(B * 751 + 255) / 256, 256, 0, stream>>>(
      h2, F(119), F(120), nullptr, nullptr, nullptr, nullptr, 1.0f, (float*)d_out,
      B, 512, 751);
}
